// HybridLayer_1597727834346
// MI455X (gfx1250) — matrix operand order repaired and checked by reading, not compiled, not run on hardware
//
#include <hip/hip_runtime.h>

// ---------------------------------------------------------------------------
// SNN conv3x3(SAME) + LIF over T=4, MI455X (gfx1250, wave32, WMMA).
//
// bf16x3 error-compensated implicit GEMM on v_wmma_f32_16x16x32_bf16:
//   y = (x_hi+x_lo)(W_hi+W_lo) ~= x_hi*W_hi + x_hi*W_lo + x_lo*W_hi
// Prep 1: x NCHW fp32 -> zero-PADDED NHWC bf16 hi/lo planes [b][58][58][128]
//         (padding removes all boundary predication from the GEMM loop; every
//          tap becomes a plain load at a compile-time immediate offset).
// Prep 2: W [cout][cin][3][3] -> [k][cout][cin] bf16 hi/lo.
// Main: implicit GEMM, M=cout(128), N=flattened (b,h,w), K=cin*9=1152.
//       Fragments loaded straight from global: bf16 x (27.6MB) is L2-resident
//       on MI455X's 192MB L2. Epilogue fuses bias + 4-step LIF, writes
//       [B,T,COUT,H,W] fp32 spikes.
// ---------------------------------------------------------------------------

typedef __bf16 bf16_t;
typedef bf16_t v16bf __attribute__((ext_vector_type(16)));
typedef bf16_t v8bf  __attribute__((ext_vector_type(8)));
typedef float  v8f   __attribute__((ext_vector_type(8)));

#define NB   16
#define NC   128
#define NH   56
#define NW   56
#define NHP  58                       // padded spatial extent
#define NHW  (NH * NW)                // 3136
#define NPOS (NB * NHW)               // 50176
#define XPAD ((size_t)NB * NHP * NHP * NC)   // 6,889,472 bf16 per padded plane
#define WPLANE ((size_t)9 * NC * NC)         // 147,456 bf16 per plane

#define WMMA_BF16(A, Bm, C) \
  __builtin_amdgcn_wmma_f32_16x16x32_bf16(false, (A), false, (Bm), (short)0, (C), false, false)

// A-fragment (16x32 bf16, MxK): lane l holds row M=l&15, K=(l>>4)*16 + {0..15}
__device__ __forceinline__ v16bf load_a_frag(const bf16_t* p) {
  v8bf lo = *(const v8bf*)(p);
  v8bf hi = *(const v8bf*)(p + 8);
  return __builtin_shufflevector(lo, hi, 0,1,2,3,4,5,6,7,8,9,10,11,12,13,14,15);
}

// ---------------------------------------------------------------------------
// Prep 1: x[b][c][h][w] fp32 -> padded NHWC bf16 hi/lo. Block per (b,h);
// writes padded row hp=h+1 including zeroed wp=0 / wp=57 columns.
// ---------------------------------------------------------------------------
__global__ __launch_bounds__(256) void prep_x(const float* __restrict__ x,
                                              bf16_t* __restrict__ xhi,
                                              bf16_t* __restrict__ xlo) {
  __shared__ float tile[NC * NW];          // [c][w], 28KB
  const int b = blockIdx.x / NH;
  const int h = blockIdx.x % NH;
  const float* src = x + ((size_t)b * NC) * NHW + (size_t)h * NW;
  for (int i = threadIdx.x; i < NC * NW; i += 256) {
    int c = i / NW, w = i - c * NW;
    tile[i] = src[(size_t)c * NHW + w];    // coalesced over w
  }
  __syncthreads();
  const size_t row = (((size_t)b * NHP) + (h + 1)) * NHP * NC;  // hp = h+1, wp = 0
  bf16_t* dh = xhi + row;
  bf16_t* dl = xlo + row;
  for (int j = threadIdx.x; j < NHP * NC; j += 256) {  // j = wp*128 + c, contiguous
    int wp = j >> 7, c = j & 127;
    float v = 0.0f;
    if (wp >= 1 && wp <= NW) v = tile[c * NW + (wp - 1)];
    bf16_t hh = (bf16_t)v;
    dh[j] = hh;
    dl[j] = (bf16_t)(v - (float)hh);
  }
}

// Zero the hp=0 and hp=57 border rows of both planes.
__global__ __launch_bounds__(256) void zero_border(bf16_t* __restrict__ xhi,
                                                   bf16_t* __restrict__ xlo) {
  const int rowlen = NHP * NC;                 // 7424
  int t = blockIdx.x * 256 + threadIdx.x;
  const int total = NB * 2 * rowlen;           // 237,568
  if (t >= total) return;
  int b   = t / (2 * rowlen);
  int r   = t - b * 2 * rowlen;
  int hi_row = r / rowlen;                     // 0 or 1
  int idx = r - hi_row * rowlen;
  int hp  = hi_row ? (NHP - 1) : 0;
  size_t off = (((size_t)b * NHP) + hp) * NHP * NC + idx;
  xhi[off] = (bf16_t)0.0f;
  xlo[off] = (bf16_t)0.0f;
}

// ---------------------------------------------------------------------------
// Prep 2: W[cout][cin][kh][kw] fp32 -> whi/wlo [k][cout][cin] bf16
// ---------------------------------------------------------------------------
__global__ __launch_bounds__(256) void prep_w(const float* __restrict__ W,
                                              bf16_t* __restrict__ whi,
                                              bf16_t* __restrict__ wlo) {
  int t = blockIdx.x * 256 + threadIdx.x;
  if (t >= (int)WPLANE) return;
  int cin  = t & 127;
  int r    = t >> 7;
  int cout = r & 127;
  int kk   = r >> 7;                           // kh*3+kw
  float v = W[((size_t)cout * NC + cin) * 9 + kk];
  bf16_t hh = (bf16_t)v;
  whi[t] = hh;
  wlo[t] = (bf16_t)(v - (float)hh);
}

// ---------------------------------------------------------------------------
// Main: implicit-GEMM conv + fused LIF.
// Block = 256 thr = 8 waves; wave w: couts [16w,16w+16); block: 64 positions.
// Per wave: 4 accumulators (16 couts x 4x16 positions). K-loop = 9 taps x 4
// cin-chunks of 32 -> 36 iters x 12 WMMA, all loads at immediate offsets from
// one precomputed base address per operand (padded layout -> no predication).
// ---------------------------------------------------------------------------
__global__ __launch_bounds__(256, 1) void conv3x3_lif_wmma(
    const bf16_t* __restrict__ xhi, const bf16_t* __restrict__ xlo,
    const bf16_t* __restrict__ whi, const bf16_t* __restrict__ wlo,
    const float* __restrict__ bias, float* __restrict__ out) {
  const int lane  = threadIdx.x & 31;
  const int wavei = threadIdx.x >> 5;
  const int cout0 = wavei << 4;            // 0..112
  const int col   = lane & 15;             // N (and A-row M) index within tile
  const int khalf = lane >> 4;             // which K half this lane holds
  const int n0    = blockIdx.x * 64;       // flattened (b,h,w) base

  // Per-lane padded base address per N-subtile: points at padded (ph, pw),
  // tap (kh,kw) lives at constant offset (kh*58+kw)*128.
  int pb[4], ph[4], pw[4];
  const bf16_t* bhp[4];
  const bf16_t* blp[4];
#pragma unroll
  for (int i = 0; i < 4; ++i) {
    int n = n0 + i * 16 + col;
    pb[i] = n / NHW;
    int hw = n - pb[i] * NHW;
    ph[i] = hw / NW;
    pw[i] = hw - ph[i] * NW;
    size_t base = ((((size_t)pb[i] * NHP) + ph[i]) * NHP + pw[i]) * NC + khalf * 16;
    bhp[i] = xhi + base;
    blp[i] = xlo + base;
  }
  // Per-lane A base: row cout0+col, K-half khalf; tap kk at offset kk*128*128.
  const bf16_t* aph = whi + ((size_t)(cout0 + col)) * NC + khalf * 16;
  const bf16_t* apl = wlo + ((size_t)(cout0 + col)) * NC + khalf * 16;

  v8f acc[4] = {{}, {}, {}, {}};

#pragma unroll
  for (int kk = 0; kk < 9; ++kk) {
    const int toff = ((kk / 3) * NHP + (kk % 3)) * NC;   // compile-time per kk
    const int aoff = kk * NC * NC;
#pragma unroll
    for (int cc = 0; cc < 4; ++cc) {
      const int c0 = cc * 32;
      v16bf Ah = load_a_frag(aph + aoff + c0);
      v16bf Al = load_a_frag(apl + aoff + c0);

      v16bf Bh[4], Bl[4];
#pragma unroll
      for (int i = 0; i < 4; ++i) {
        Bh[i] = *(const v16bf*)(bhp[i] + toff + c0);
        Bl[i] = *(const v16bf*)(blp[i] + toff + c0);
      }
      // Interleaved so each accumulator is touched every 4th WMMA (hazard room).
#pragma unroll
      for (int i = 0; i < 4; ++i) acc[i] = WMMA_BF16(Ah, Bh[i], acc[i]);
#pragma unroll
      for (int i = 0; i < 4; ++i) acc[i] = WMMA_BF16(Ah, Bl[i], acc[i]);
#pragma unroll
      for (int i = 0; i < 4; ++i) acc[i] = WMMA_BF16(Al, Bh[i], acc[i]);
    }
  }

  // Epilogue: D VGPR r -> cout = cout0 + khalf*8 + r, N = col.
  v8f bb = *(const v8f*)(bias + cout0 + khalf * 8);

#pragma unroll
  for (int i = 0; i < 4; ++i) {
    const int hw = ph[i] * NW + pw[i];
#pragma unroll
    for (int r = 0; r < 8; ++r) {
      const int coutr = cout0 + khalf * 8 + r;
      float y   = acc[i][r] + bb[r];
      float mem = 0.0f;
      size_t base = ((size_t)pb[i] * 4 * NC + coutr) * NHW + hw;
#pragma unroll
      for (int t = 0; t < 4; ++t) {
        mem += y;
        float s = (mem >= 1.0f) ? 1.0f : 0.0f;   // THRESHOLD = 1.0
        mem -= s;
        out[base + (size_t)t * NC * NHW] = s;    // lanes 0-15 coalesce over w
      }
    }
  }
}

// ---------------------------------------------------------------------------
// kernel_launch
// inputs: d_in[0]=x fp32 [16,128,56,56], d_in[1]=W fp32 [128,128,3,3],
//         d_in[2]=b fp32 [128]; d_out fp32 [16,4,128,56,56]
// workspace: 2 padded x-planes + 2 W-planes of bf16 (~28.1 MB)
// ---------------------------------------------------------------------------
extern "C" void kernel_launch(void* const* d_in, const int* in_sizes, int n_in,
                              void* d_out, int out_size, void* d_ws, size_t ws_size,
                              hipStream_t stream) {
  const float* x = (const float*)d_in[0];
  const float* W = (const float*)d_in[1];
  const float* b = (const float*)d_in[2];
  float* out = (float*)d_out;

  bf16_t* xhi = (bf16_t*)d_ws;
  bf16_t* xlo = xhi + XPAD;
  bf16_t* whi = xlo + XPAD;
  bf16_t* wlo = whi + WPLANE;

  zero_border<<<(NB * 2 * NHP * NC + 255) / 256, 256, 0, stream>>>(xhi, xlo);
  prep_x<<<NB * NH, 256, 0, stream>>>(x, xhi, xlo);
  prep_w<<<(int)((WPLANE + 255) / 256), 256, 0, stream>>>(W, whi, wlo);
  conv3x3_lif_wmma<<<NPOS / 64, 256, 0, stream>>>(xhi, xlo, whi, wlo, b, out);
}
